// Struct2SeqAO_83700322664796
// MI455X (gfx1250) — matrix operand order repaired and checked by reading, not compiled, not run on hardware
//
#include <hip/hip_runtime.h>

// ---------------------------------------------------------------------------
// Struct2Seq forward on MI455X (gfx1250): bf16 WMMA GEMMs + fp32 glue.
// Round 3: fuse neighbor-concat into the K/V projection GEMMs (gemm_kv) so the
// h_EV tensor (126-189 MB/layer) is never materialized; drop the SYS-scope
// prefetch that bloated the inner loop.
// ---------------------------------------------------------------------------

#define BB 16
#define NN 512
#define KNB 30
#define HH 128
#define NHEAD 4
#define DH 32
#define VOCAB 20

typedef __attribute__((ext_vector_type(16))) __bf16 v16bf;
typedef __attribute__((ext_vector_type(8)))  float  v8f;

union Frag16 { v16bf v; uint4 q[2]; unsigned int u[8]; };

static __device__ __forceinline__ unsigned short f2bf(float x) {
    unsigned int u = __float_as_uint(x);
    u = (u + 0x7FFFu + ((u >> 16) & 1u)) >> 16;
    return (unsigned short)u;
}
static __device__ __forceinline__ float bf2f(unsigned short h) {
    return __uint_as_float(((unsigned int)h) << 16);
}
static __device__ __forceinline__ unsigned int pk2(float a, float b) {
    return (unsigned int)f2bf(a) | ((unsigned int)f2bf(b) << 16);
}

#define NEGINF (-3.402823466e38f)

// ---------------------------------------------------------------------------
// Weight prep: f32 (Kd x Nd) row-major  ->  bf16 (Nd x Kd) row-major
// ---------------------------------------------------------------------------
__global__ void transpose_bf16(const float* __restrict__ in,
                               unsigned short* __restrict__ out,
                               int Kd, int Nd) {
    int t = blockIdx.x * 256 + threadIdx.x;
    if (t >= Kd * Nd) return;
    int k = t / Nd, n = t % Nd;
    out[(size_t)n * Kd + k] = f2bf(in[t]);
}

// ---------------------------------------------------------------------------
// Fragment loaders (16-byte aligned by construction: Kd % 32 == 0)
// A 16x32 bf16 layout: lane m=l&15, g=l>>4; elems 0..7 -> k=g*8+0..7,
//                      elems 8..15 -> k=16+g*8+0..7
// B 32x16 bf16 layout: lane n=l&15, g=l>>4; elems 0..7 -> k=g*8+0..7,
//                      elems 8..15 -> k=16+g*8+0..7
// ---------------------------------------------------------------------------
static __device__ __forceinline__ void loadA(Frag16& f, const unsigned short* Arow,
                                             int kb, int g) {
    f.q[0] = *reinterpret_cast<const uint4*>(Arow + kb + g * 8);
    f.q[1] = *reinterpret_cast<const uint4*>(Arow + kb + 16 + g * 8);
}
static __device__ __forceinline__ void loadB(Frag16& f, const unsigned short* Brow,
                                             int kb, int g) {
    f.q[0] = *reinterpret_cast<const uint4*>(Brow + kb + g * 8);
    f.q[1] = *reinterpret_cast<const uint4*>(Brow + kb + 16 + g * 8);
}

// Pack 16 f32 (two aligned float4 pairs) * scale -> bf16 fragment
static __device__ __forceinline__ void packA_f32(Frag16& f, const float* p, int g,
                                                 float s) {
    float4 x0 = *reinterpret_cast<const float4*>(p + g * 8);
    float4 x1 = *reinterpret_cast<const float4*>(p + g * 8 + 4);
    float4 x2 = *reinterpret_cast<const float4*>(p + 16 + g * 8);
    float4 x3 = *reinterpret_cast<const float4*>(p + 16 + g * 8 + 4);
    f.u[0] = pk2(x0.x * s, x0.y * s);
    f.u[1] = pk2(x0.z * s, x0.w * s);
    f.u[2] = pk2(x1.x * s, x1.y * s);
    f.u[3] = pk2(x1.z * s, x1.w * s);
    f.u[4] = pk2(x2.x * s, x2.y * s);
    f.u[5] = pk2(x2.z * s, x2.w * s);
    f.u[6] = pk2(x3.x * s, x3.y * s);
    f.u[7] = pk2(x3.z * s, x3.w * s);
}
// Blend two f32 rows: sb*pb + sc*pc -> bf16 fragment
static __device__ __forceinline__ void packA_blend(Frag16& f, const float* pb,
                                                   const float* pc, int g,
                                                   float sb, float sc) {
    float4 b0 = *reinterpret_cast<const float4*>(pb + g * 8);
    float4 b1 = *reinterpret_cast<const float4*>(pb + g * 8 + 4);
    float4 b2 = *reinterpret_cast<const float4*>(pb + 16 + g * 8);
    float4 b3 = *reinterpret_cast<const float4*>(pb + 16 + g * 8 + 4);
    float4 c0 = *reinterpret_cast<const float4*>(pc + g * 8);
    float4 c1 = *reinterpret_cast<const float4*>(pc + g * 8 + 4);
    float4 c2 = *reinterpret_cast<const float4*>(pc + 16 + g * 8);
    float4 c3 = *reinterpret_cast<const float4*>(pc + 16 + g * 8 + 4);
    f.u[0] = pk2(sb * b0.x + sc * c0.x, sb * b0.y + sc * c0.y);
    f.u[1] = pk2(sb * b0.z + sc * c0.z, sb * b0.w + sc * c0.w);
    f.u[2] = pk2(sb * b1.x + sc * c1.x, sb * b1.y + sc * c1.y);
    f.u[3] = pk2(sb * b1.z + sc * c1.z, sb * b1.w + sc * c1.w);
    f.u[4] = pk2(sb * b2.x + sc * c2.x, sb * b2.y + sc * c2.y);
    f.u[5] = pk2(sb * b2.z + sc * c2.z, sb * b2.w + sc * c2.w);
    f.u[6] = pk2(sb * b3.x + sc * c3.x, sb * b3.y + sc * c3.y);
    f.u[7] = pk2(sb * b3.z + sc * c3.z, sb * b3.w + sc * c3.w);
}

// ---------------------------------------------------------------------------
// Generic GEMM: C[M,N] = A[M,Kd](bf16) * Wt[N,Kd]^T(bf16) (+bias)(+relu)
// block = 256 threads = 8 waves; block tile 128m x 64n; wave tile 16m x 64n.
// Software pipelined: frag set for step k+1 loads while step k's 4 WMMAs run.
// ---------------------------------------------------------------------------
__global__ __launch_bounds__(256) void gemm_bf16(
    const unsigned short* __restrict__ A,
    const unsigned short* __restrict__ Wt,
    const float* __restrict__ bias,
    float* __restrict__ Cf,
    unsigned short* __restrict__ Cbf,
    int M, int Nn, int Kd, int relu)
{
    const int wave = threadIdx.x >> 5;
    const int lane = threadIdx.x & 31;
    const int m0 = blockIdx.x * 128 + wave * 16;
    const int n0 = blockIdx.y * 64;
    const int ml = lane & 15;
    const int g  = lane >> 4;

    v8f acc0 = {}, acc1 = {}, acc2 = {}, acc3 = {};

    const unsigned short* Arow  = A  + (size_t)(m0 + ml) * Kd;
    const unsigned short* Brow0 = Wt + (size_t)(n0 +  0 + ml) * Kd;
    const unsigned short* Brow1 = Wt + (size_t)(n0 + 16 + ml) * Kd;
    const unsigned short* Brow2 = Wt + (size_t)(n0 + 32 + ml) * Kd;
    const unsigned short* Brow3 = Wt + (size_t)(n0 + 48 + ml) * Kd;

    Frag16 aX, b0X, b1X, b2X, b3X;   // ping
    Frag16 aY, b0Y, b1Y, b2Y, b3Y;   // pong

    loadA(aX,  Arow,  0, g);
    loadB(b0X, Brow0, 0, g);
    loadB(b1X, Brow1, 0, g);
    loadB(b2X, Brow2, 0, g);
    loadB(b3X, Brow3, 0, g);

    int kb = 0;
    for (;;) {
        {   // even step: consume X, preload Y
            int kn = kb + 32;
            if (kn < Kd) {
                loadA(aY,  Arow,  kn, g);
                loadB(b0Y, Brow0, kn, g);
                loadB(b1Y, Brow1, kn, g);
                loadB(b2Y, Brow2, kn, g);
                loadB(b3Y, Brow3, kn, g);
            }
            acc0 = __builtin_amdgcn_wmma_f32_16x16x32_bf16(false, aX.v, false, b0X.v, (short)0, acc0, false, false);
            acc1 = __builtin_amdgcn_wmma_f32_16x16x32_bf16(false, aX.v, false, b1X.v, (short)0, acc1, false, false);
            acc2 = __builtin_amdgcn_wmma_f32_16x16x32_bf16(false, aX.v, false, b2X.v, (short)0, acc2, false, false);
            acc3 = __builtin_amdgcn_wmma_f32_16x16x32_bf16(false, aX.v, false, b3X.v, (short)0, acc3, false, false);
            kb += 32;
            if (kb >= Kd) break;
        }
        {   // odd step: consume Y, preload X
            int kn = kb + 32;
            if (kn < Kd) {
                loadA(aX,  Arow,  kn, g);
                loadB(b0X, Brow0, kn, g);
                loadB(b1X, Brow1, kn, g);
                loadB(b2X, Brow2, kn, g);
                loadB(b3X, Brow3, kn, g);
            }
            acc0 = __builtin_amdgcn_wmma_f32_16x16x32_bf16(false, aY.v, false, b0Y.v, (short)0, acc0, false, false);
            acc1 = __builtin_amdgcn_wmma_f32_16x16x32_bf16(false, aY.v, false, b1Y.v, (short)0, acc1, false, false);
            acc2 = __builtin_amdgcn_wmma_f32_16x16x32_bf16(false, aY.v, false, b2Y.v, (short)0, acc2, false, false);
            acc3 = __builtin_amdgcn_wmma_f32_16x16x32_bf16(false, aY.v, false, b3Y.v, (short)0, acc3, false, false);
            kb += 32;
            if (kb >= Kd) break;
        }
    }

    const v8f* accs[4] = { &acc0, &acc1, &acc2, &acc3 };
#pragma unroll
    for (int t = 0; t < 4; ++t) {
        int n = n0 + t * 16 + ml;
        float bvv = bias ? bias[n] : 0.f;
#pragma unroll
        for (int r = 0; r < 8; ++r) {
            int m = m0 + r + g * 8;
            float x = (*accs[t])[r] + bvv;
            if (relu) x = x > 0.f ? x : 0.f;
            size_t idx = (size_t)m * Nn + n;
            if (Cf)  Cf[idx]  = x;
            if (Cbf) Cbf[idx] = f2bf(x);
        }
    }
}

// ---------------------------------------------------------------------------
// Fused concat + K/V projection GEMM over edge rows.
//  A row r (edge b,i,k):  enc (dec=0, Kd=256): [hE | cvt(hV[j])]
//                         dec (dec=1, Kd=384): [hEmask | bw*hS[j] | bw*hV[j]+fw*hVenc[j]]
//  (hE passed in pre-scaled by mask for dec). Node tensors are 4 MB and
//  L2-resident (reused 30x) -> gathers never touch HBM.
// ---------------------------------------------------------------------------
__global__ __launch_bounds__(256) void gemm_kv(
    const unsigned short* __restrict__ hE,
    const float* __restrict__ nA,
    const float* __restrict__ nB,
    const float* __restrict__ nC,
    const int* __restrict__ Eidx,
    const float* __restrict__ mask,
    const unsigned short* __restrict__ Wt,
    unsigned short* __restrict__ Cbf,
    int Kd, int dec)
{
    const int wave = threadIdx.x >> 5;
    const int lane = threadIdx.x & 31;
    const int m0 = blockIdx.x * 128 + wave * 16;
    const int n0 = blockIdx.y * 64;
    const int ml = lane & 15;
    const int g  = lane >> 4;

    const int r = m0 + ml;                 // edge row
    const int j = Eidx[r];
    const int i = (r / KNB) & (NN - 1);
    const int b = r / (KNB * NN);
    const float m = mask[b * NN + i];
    float s1 = 1.f, sb = 0.f, sc = 0.f;
    if (dec) {
        float ar = (j < i) ? 1.f : 0.f;
        sb = m * ar;          // bw
        sc = m - sb;          // fw
        s1 = sb;              // seg1 scale (bw * hS)
    }
    const unsigned short* hErow = hE + (size_t)r * HH;
    const float* nArow = nA + (size_t)(b * NN + j) * HH;
    const float* nBrow = nB ? nB + (size_t)(b * NN + j) * HH : nullptr;
    const float* nCrow = nC ? nC + (size_t)(b * NN + j) * HH : nullptr;

    const unsigned short* Brow0 = Wt + (size_t)(n0 +  0 + ml) * Kd;
    const unsigned short* Brow1 = Wt + (size_t)(n0 + 16 + ml) * Kd;
    const unsigned short* Brow2 = Wt + (size_t)(n0 + 32 + ml) * Kd;
    const unsigned short* Brow3 = Wt + (size_t)(n0 + 48 + ml) * Kd;

    auto loadAkv = [&](Frag16& f, int kb) {
        if (kb < HH) {
            f.q[0] = *reinterpret_cast<const uint4*>(hErow + kb + g * 8);
            f.q[1] = *reinterpret_cast<const uint4*>(hErow + kb + 16 + g * 8);
        } else if (!dec || kb < 2 * HH) {
            packA_f32(f, nArow + (kb - HH), g, s1);
        } else {
            packA_blend(f, nBrow + (kb - 2 * HH), nCrow + (kb - 2 * HH), g, sb, sc);
        }
    };

    v8f acc0 = {}, acc1 = {}, acc2 = {}, acc3 = {};
    Frag16 aX, b0X, b1X, b2X, b3X;
    Frag16 aY, b0Y, b1Y, b2Y, b3Y;

    loadAkv(aX, 0);
    loadB(b0X, Brow0, 0, g);
    loadB(b1X, Brow1, 0, g);
    loadB(b2X, Brow2, 0, g);
    loadB(b3X, Brow3, 0, g);

    int kb = 0;
    for (;;) {
        {
            int kn = kb + 32;
            if (kn < Kd) {
                loadAkv(aY, kn);
                loadB(b0Y, Brow0, kn, g);
                loadB(b1Y, Brow1, kn, g);
                loadB(b2Y, Brow2, kn, g);
                loadB(b3Y, Brow3, kn, g);
            }
            acc0 = __builtin_amdgcn_wmma_f32_16x16x32_bf16(false, aX.v, false, b0X.v, (short)0, acc0, false, false);
            acc1 = __builtin_amdgcn_wmma_f32_16x16x32_bf16(false, aX.v, false, b1X.v, (short)0, acc1, false, false);
            acc2 = __builtin_amdgcn_wmma_f32_16x16x32_bf16(false, aX.v, false, b2X.v, (short)0, acc2, false, false);
            acc3 = __builtin_amdgcn_wmma_f32_16x16x32_bf16(false, aX.v, false, b3X.v, (short)0, acc3, false, false);
            kb += 32;
            if (kb >= Kd) break;
        }
        {
            int kn = kb + 32;
            if (kn < Kd) {
                loadAkv(aX, kn);
                loadB(b0X, Brow0, kn, g);
                loadB(b1X, Brow1, kn, g);
                loadB(b2X, Brow2, kn, g);
                loadB(b3X, Brow3, kn, g);
            }
            acc0 = __builtin_amdgcn_wmma_f32_16x16x32_bf16(false, aY.v, false, b0Y.v, (short)0, acc0, false, false);
            acc1 = __builtin_amdgcn_wmma_f32_16x16x32_bf16(false, aY.v, false, b1Y.v, (short)0, acc1, false, false);
            acc2 = __builtin_amdgcn_wmma_f32_16x16x32_bf16(false, aY.v, false, b2Y.v, (short)0, acc2, false, false);
            acc3 = __builtin_amdgcn_wmma_f32_16x16x32_bf16(false, aY.v, false, b3Y.v, (short)0, acc3, false, false);
            kb += 32;
            if (kb >= Kd) break;
        }
    }

    const v8f* accs[4] = { &acc0, &acc1, &acc2, &acc3 };
#pragma unroll
    for (int t = 0; t < 4; ++t) {
        int n = n0 + t * 16 + ml;
#pragma unroll
        for (int rr = 0; rr < 8; ++rr) {
            int mm = m0 + rr + g * 8;
            Cbf[(size_t)mm * HH + n] = f2bf((*accs[t])[rr]);
        }
    }
}

// ---------------------------------------------------------------------------
// hEmask = mask[b,i] * hE   (decoder's pre-scaled edge embedding)
// ---------------------------------------------------------------------------
__global__ void scale_hE(const unsigned short* __restrict__ in,
                         const float* __restrict__ mask,
                         unsigned short* __restrict__ out) {
    int t = blockIdx.x * 256 + threadIdx.x;
    if (t >= BB * NN * KNB * HH) return;
    int r = t >> 7;
    int i = (r / KNB) & (NN - 1);
    int b = r / (KNB * NN);
    out[t] = f2bf(mask[b * NN + i] * bf2f(in[t]));
}

// ---------------------------------------------------------------------------
// Distances + stable top-K selection (K smallest D_adjust, ties -> lower idx)
// ---------------------------------------------------------------------------
__global__ __launch_bounds__(128) void dist_topk(
    const float* __restrict__ X, const float* __restrict__ mask,
    int* __restrict__ Eidx, float* __restrict__ Dnb)
{
    __shared__ float d[NN];
    __shared__ float rv[128];
    __shared__ int   ri[128];
    const int r = blockIdx.x;          // b*N + i
    const int b = r >> 9;
    const int tid = threadIdx.x;

    const float* xi = X + ((size_t)r * 4 + 1) * 3;
    const float xi0 = xi[0], xi1 = xi[1], xi2 = xi[2];
    const float mi = mask[r];

    for (int j = tid; j < NN; j += 128) {
        const float* xj = X + (((size_t)(b * NN + j)) * 4 + 1) * 3;
        float dx = xi0 - xj[0], dy = xi1 - xj[1], dz = xi2 - xj[2];
        float mj = mask[b * NN + j];
        d[j] = mi * mj * sqrtf(dx * dx + dy * dy + dz * dz + 1e-6f);
    }
    __syncthreads();

    float mx = 0.f;
    for (int j = tid; j < NN; j += 128) mx = fmaxf(mx, d[j]);
    rv[tid] = mx; __syncthreads();
    for (int s = 64; s; s >>= 1) {
        if (tid < s) rv[tid] = fmaxf(rv[tid], rv[tid + s]);
        __syncthreads();
    }
    const float Dmax = rv[0];
    __syncthreads();

    for (int j = tid; j < NN; j += 128) {
        float mj = mask[b * NN + j];
        d[j] += (1.f - mi * mj) * Dmax;
    }
    __syncthreads();

    for (int kk = 0; kk < KNB; ++kk) {
        float bv = 3.0e38f; int bi = NN;
        for (int j = tid; j < NN; j += 128) {
            if (d[j] < bv) { bv = d[j]; bi = j; }
        }
        rv[tid] = bv; ri[tid] = bi; __syncthreads();
        for (int s = 64; s; s >>= 1) {
            if (tid < s) {
                if (rv[tid + s] < rv[tid] ||
                    (rv[tid + s] == rv[tid] && ri[tid + s] < ri[tid])) {
                    rv[tid] = rv[tid + s]; ri[tid] = ri[tid + s];
                }
            }
            __syncthreads();
        }
        if (tid == 0) {
            Eidx[(size_t)r * KNB + kk] = ri[0];
            Dnb [(size_t)r * KNB + kk] = rv[0];
            d[ri[0]] = 3.0e38f;
        }
        __syncthreads();
    }
}

// ---------------------------------------------------------------------------
// Dihedral features -> V6 (B,N,6)
// ---------------------------------------------------------------------------
static __device__ __forceinline__ void norm3(float* v) {
    float n = sqrtf(v[0]*v[0] + v[1]*v[1] + v[2]*v[2]) + 1e-7f;
    v[0] /= n; v[1] /= n; v[2] /= n;
}
static __device__ __forceinline__ void cross3(const float* a, const float* b, float* o) {
    o[0] = a[1]*b[2] - a[2]*b[1];
    o[1] = a[2]*b[0] - a[0]*b[2];
    o[2] = a[0]*b[1] - a[1]*b[0];
}
static __device__ __forceinline__ float dot3(const float* a, const float* b) {
    return a[0]*b[0] + a[1]*b[1] + a[2]*b[2];
}

__global__ void dihedral(const float* __restrict__ X, float* __restrict__ V6) {
    int t = blockIdx.x * blockDim.x + threadIdx.x;
    const int total = BB * 3 * NN;
    if (t >= total) return;
    int b = t / (3 * NN);
    int p = t % (3 * NN);
    float ang = 0.f;
    int q = p - 1;
    if (q >= 0 && q < 3 * NN - 3) {
        float a[4][3];
#pragma unroll
        for (int s = 0; s < 4; ++s) {
            int m = q + s;
            const float* xp = X + (((size_t)(b * NN + m / 3)) * 4 + (m % 3)) * 3;
            a[s][0] = xp[0]; a[s][1] = xp[1]; a[s][2] = xp[2];
        }
        float u2[3], u1[3], u0[3];
#pragma unroll
        for (int c = 0; c < 3; ++c) {
            u2[c] = a[1][c] - a[0][c];
            u1[c] = a[2][c] - a[1][c];
            u0[c] = a[3][c] - a[2][c];
        }
        norm3(u2); norm3(u1); norm3(u0);
        float n2[3], n1[3];
        cross3(u2, u1, n2); norm3(n2);
        cross3(u1, u0, n1); norm3(n1);
        float cosD = dot3(n2, n1);
        cosD = fminf(fmaxf(cosD, -1.f + 1e-7f), 1.f - 1e-7f);
        float s = dot3(u2, n1);
        float sg = (s > 0.f) ? 1.f : ((s < 0.f) ? -1.f : 0.f);
        ang = sg * acosf(cosD);
    }
    int n = p / 3, c = p % 3;
    V6[((size_t)(b * NN + n)) * 6 + c]     = cosf(ang);
    V6[((size_t)(b * NN + n)) * 6 + 3 + c] = sinf(ang);
}

// ---------------------------------------------------------------------------
// Edge features (pos-enc 16 + rbf 16) -> bf16 (B,N,K,32)
// ---------------------------------------------------------------------------
__global__ void edge_feat(const int* __restrict__ Eidx,
                          const float* __restrict__ Dnb,
                          unsigned short* __restrict__ out) {
    int t = blockIdx.x * 256 + threadIdx.x;
    if (t >= BB * NN * KNB) return;
    int i = (t / KNB) & (NN - 1);
    float d = (float)(Eidx[t] - i);
    float D = Dnb[t];
    unsigned short* o = out + (size_t)t * 32;
#pragma unroll
    for (int f = 0; f < 8; ++f) {
        float freq = expf((float)(2 * f) * (-9.210340371976184f / 16.f));
        float ang = d * freq;
        o[f]     = f2bf(cosf(ang));
        o[8 + f] = f2bf(sinf(ang));
    }
#pragma unroll
    for (int rr = 0; rr < 16; ++rr) {
        float mu = 20.f * (float)rr / 15.f;
        float e = (D - mu) / 1.25f;
        o[16 + rr] = f2bf(expf(-e * e));
    }
}

// ---------------------------------------------------------------------------
// V6 @ Wfn + bfn  (K=6: tiny, plain VALU)
// ---------------------------------------------------------------------------
__global__ void noderaw(const float* __restrict__ V6,
                        const float* __restrict__ Wfn,
                        const float* __restrict__ bfn,
                        float* __restrict__ out) {
    int t = blockIdx.x * 256 + threadIdx.x;
    if (t >= BB * NN * HH) return;
    int r = t >> 7, h = t & 127;
    float acc = bfn[h];
#pragma unroll
    for (int c = 0; c < 6; ++c) acc += V6[(size_t)r * 6 + c] * Wfn[c * HH + h];
    out[t] = acc;
}

// ---------------------------------------------------------------------------
// Row LayerNorm over 128 (one wave/row), optional residual add, optional
// post-LN mask multiply; dual f32 / bf16 outputs.
// ---------------------------------------------------------------------------
__global__ __launch_bounds__(32) void ln128(
    const float* __restrict__ in, const float* __restrict__ add,
    const float* __restrict__ maskp,
    float* __restrict__ outf, unsigned short* __restrict__ outbf)
{
    const int r = blockIdx.x, lane = threadIdx.x;
    const size_t base = (size_t)r * HH;
    float x[4]; float s = 0.f, sq = 0.f;
#pragma unroll
    for (int j = 0; j < 4; ++j) {
        float v = in[base + lane + 32 * j];
        if (add) v += add[base + lane + 32 * j];
        x[j] = v; s += v; sq += v * v;
    }
#pragma unroll
    for (int m = 16; m; m >>= 1) {
        s  += __shfl_xor(s,  m, 32);
        sq += __shfl_xor(sq, m, 32);
    }
    float mean = s * (1.f / HH);
    float var  = sq * (1.f / HH) - mean * mean;
    float inv  = rsqrtf(var + 1e-6f);
    float mk   = maskp ? maskp[r] : 1.f;
#pragma unroll
    for (int j = 0; j < 4; ++j) {
        float y = (x[j] - mean) * inv * mk;
        if (outf)  outf [base + lane + 32 * j] = y;
        if (outbf) outbf[base + lane + 32 * j] = f2bf(y);
    }
}

// ---------------------------------------------------------------------------
// h_S = Ws[S]
// ---------------------------------------------------------------------------
__global__ void gather_hs(const int* __restrict__ S, const float* __restrict__ Ws,
                          float* __restrict__ hS) {
    int t = blockIdx.x * 256 + threadIdx.x;
    if (t >= BB * NN * HH) return;
    int r = t >> 7, h = t & 127;
    hS[t] = Ws[(size_t)S[r] * HH + h];
}

// ---------------------------------------------------------------------------
// Neighbor attention: one block per (b,i), one wave per head.
// ---------------------------------------------------------------------------
__global__ __launch_bounds__(128) void attn(
    const float* __restrict__ Q, const unsigned short* __restrict__ Kp,
    const unsigned short* __restrict__ Vp, const int* __restrict__ Eidx,
    const float* __restrict__ mask, unsigned short* __restrict__ out,
    int encmask)
{
    __shared__ float qs[HH];
    __shared__ float att[NHEAD][32];
    const int r = blockIdx.x, tid = threadIdx.x;
    qs[tid] = Q[(size_t)r * HH + tid];
    __syncthreads();

    const int h = tid >> 5, lane = tid & 31;
    float logit = NEGINF;
    float m = 1.f;
    if (lane < KNB) {
        const unsigned short* kr = Kp + ((size_t)r * KNB + lane) * HH + h * DH;
        float acc = 0.f;
#pragma unroll
        for (int t = 0; t < DH; ++t) acc += qs[h * DH + t] * bf2f(kr[t]);
        acc *= 0.17677669529663687f;   // 1/sqrt(32)
        if (encmask) {
            int j = Eidx[(size_t)r * KNB + lane];
            m = mask[r] * mask[(r & ~(NN - 1)) + j];
            logit = (m > 0.f) ? acc : NEGINF;
        } else {
            logit = acc;
        }
    }
    float mx = logit;
#pragma unroll
    for (int s = 16; s; s >>= 1) mx = fmaxf(mx, __shfl_xor(mx, s, 32));
    float e = (lane < KNB) ? expf(logit - mx) : 0.f;
    float sum = e;
#pragma unroll
    for (int s = 16; s; s >>= 1) sum += __shfl_xor(sum, s, 32);
    float p = e / sum;
    if (encmask) p *= m;
    att[h][lane] = (lane < KNB) ? p : 0.f;
    __syncthreads();

    float acc = 0.f;
    const unsigned short* vr = Vp + (size_t)r * KNB * HH + h * DH + lane;
    for (int k = 0; k < KNB; ++k) acc += att[h][k] * bf2f(vr[(size_t)k * HH]);
    out[(size_t)r * HH + tid] = f2bf(acc);
}

// ---------------------------------------------------------------------------
// Output head: logits (H->20) + log_softmax, one wave per row.
// ---------------------------------------------------------------------------
__global__ __launch_bounds__(32) void out_head(
    const float* __restrict__ hV, const float* __restrict__ Wout,
    const float* __restrict__ bout, float* __restrict__ out)
{
    const int r = blockIdx.x, lane = threadIdx.x;
    float logit = NEGINF;
    if (lane < VOCAB) {
        float acc = bout[lane];
        const float* hr = hV + (size_t)r * HH;
        for (int c = 0; c < HH; ++c) acc += hr[c] * Wout[c * VOCAB + lane];
        logit = acc;
    }
    float mx = logit;
#pragma unroll
    for (int s = 16; s; s >>= 1) mx = fmaxf(mx, __shfl_xor(mx, s, 32));
    float e = (lane < VOCAB) ? expf(logit - mx) : 0.f;
    float sum = e;
#pragma unroll
    for (int s = 16; s; s >>= 1) sum += __shfl_xor(sum, s, 32);
    if (lane < VOCAB) out[(size_t)r * VOCAB + lane] = logit - mx - logf(sum);
}

// ---------------------------------------------------------------------------
// Host orchestration
// ---------------------------------------------------------------------------
extern "C" void kernel_launch(void* const* d_in, const int* in_sizes, int n_in,
                              void* d_out, int out_size, void* d_ws, size_t ws_size,
                              hipStream_t stream) {
    (void)in_sizes; (void)n_in; (void)out_size; (void)ws_size;

    const float* X    = (const float*)d_in[0];
    const float* mask = (const float*)d_in[1];
    const int*   S    = (const int*)  d_in[2];
    const float* Wfn  = (const float*)d_in[3];
    const float* bfn  = (const float*)d_in[4];
    const float* Wfe  = (const float*)d_in[5];
    const float* bfe  = (const float*)d_in[6];
    const float* Wv   = (const float*)d_in[7];
    const float* bv   = (const float*)d_in[8];
    const float* We   = (const float*)d_in[9];
    const float* be   = (const float*)d_in[10];
    const float* Ws_  = (const float*)d_in[11];
    const float* eWQ  = (const float*)d_in[12];
    const float* eWK  = (const float*)d_in[13];
    const float* eWV  = (const float*)d_in[14];
    const float* eWO  = (const float*)d_in[15];
    const float* eW1  = (const float*)d_in[16];
    const float* eb1  = (const float*)d_in[17];
    const float* eW2  = (const float*)d_in[18];
    const float* eb2  = (const float*)d_in[19];
    const float* dWQ  = (const float*)d_in[20];
    const float* dWK  = (const float*)d_in[21];
    const float* dWV  = (const float*)d_in[22];
    const float* dWO  = (const float*)d_in[23];
    const float* dW1  = (const float*)d_in[24];
    const float* db1  = (const float*)d_in[25];
    const float* dW2  = (const float*)d_in[26];
    const float* db2  = (const float*)d_in[27];
    const float* Wout = (const float*)d_in[28];
    const float* bout = (const float*)d_in[29];

    const int RV = BB * NN;          // 8192 node rows
    const int RE = BB * NN * KNB;    // 245760 edge rows

    size_t off = 0;
    auto alloc = [&](size_t bytes) -> void* {
        off = (off + 255) & ~(size_t)255;
        void* p = (char*)d_ws + off;
        off += bytes;
        return p;
    };

    auto prep = [&](const float* w, int Kd, int Nd) -> unsigned short* {
        unsigned short* wt = (unsigned short*)alloc((size_t)Kd * Nd * 2);
        int tot = Kd * Nd;
        transpose_bf16<<<dim3((tot + 255) / 256), dim3(256), 0, stream>>>(w, wt, Kd, Nd);
        return wt;
    };

    // ---- weight prep (bf16, transposed to N x K) ----
    unsigned short* Wfe_t = prep(Wfe, 32, HH);
    unsigned short* We_t  = prep(We,  HH, HH);
    unsigned short* Wv_t  = prep(Wv,  HH, HH);
    unsigned short *eWQt[3], *eWKt[3], *eWVt[3], *eWOt[3], *eW1t[3], *eW2t[3];
    unsigned short *dWQt[3], *dWKt[3], *dWVt[3], *dWOt[3], *dW1t[3], *dW2t[3];
    for (int l = 0; l < 3; ++l) {
        eWQt[l] = prep(eWQ + (size_t)l * HH * HH,       HH,     HH);
        eWKt[l] = prep(eWK + (size_t)l * 2 * HH * HH,   2 * HH, HH);
        eWVt[l] = prep(eWV + (size_t)l * 2 * HH * HH,   2 * HH, HH);
        eWOt[l] = prep(eWO + (size_t)l * HH * HH,       HH,     HH);
        eW1t[l] = prep(eW1 + (size_t)l * HH * 4 * HH,   HH,     4 * HH);
        eW2t[l] = prep(eW2 + (size_t)l * 4 * HH * HH,   4 * HH, HH);
    }
    for (int l = 0; l < 3; ++l) {
        dWQt[l] = prep(dWQ + (size_t)l * HH * HH,       HH,     HH);
        dWKt[l] = prep(dWK + (size_t)l * 3 * HH * HH,   3 * HH, HH);
        dWVt[l] = prep(dWV + (size_t)l * 3 * HH * HH,   3 * HH, HH);
        dWOt[l] = prep(dWO + (size_t)l * HH * HH,       HH,     HH);
        dW1t[l] = prep(dW1 + (size_t)l * HH * 4 * HH,   HH,     4 * HH);
        dW2t[l] = prep(dW2 + (size_t)l * 4 * HH * HH,   4 * HH, HH);
    }

    // ---- activations / scratch ----
    int*   Eidx    = (int*)  alloc((size_t)RE * 4);
    float* Dnb     = (float*)alloc((size_t)RE * 4);
    float* V6      = (float*)alloc((size_t)RV * 6 * 4);
    float* vraw    = (float*)alloc((size_t)RV * HH * 4);
    unsigned short* Efeat = (unsigned short*)alloc((size_t)RE * 32 * 2);
    unsigned short* hEbf   = (unsigned short*)alloc((size_t)RE * HH * 2);
    unsigned short* hEmask = (unsigned short*)alloc((size_t)RE * HH * 2);
    float* hV      = (float*)alloc((size_t)RV * HH * 4);
    unsigned short* hVbf  = (unsigned short*)alloc((size_t)RV * HH * 2);
    float* hVenc   = (float*)alloc((size_t)RV * HH * 4);
    float* hS      = (float*)alloc((size_t)RV * HH * 4);
    float* Ep      = (float*)alloc((size_t)RE * HH * 4);                 // f32 E proj
    unsigned short* Kp    = (unsigned short*)alloc((size_t)RE * HH * 2); // also Eln
    unsigned short* Vp    = (unsigned short*)alloc((size_t)RE * HH * 2);
    float* Qb      = (float*)alloc((size_t)RV * HH * 4);
    unsigned short* attno = (unsigned short*)alloc((size_t)RV * HH * 2); // also Vln
    float* dhb     = (float*)alloc((size_t)RV * HH * 4);
    unsigned short* ffn   = (unsigned short*)alloc((size_t)RV * 4 * HH * 2);

    auto gemm = [&](const unsigned short* A, const unsigned short* Wt,
                    const float* bias, float* Cf, unsigned short* Cbf,
                    int M, int Nn, int Kd, int relu) {
        gemm_bf16<<<dim3(M / 128, Nn / 64), dim3(256), 0, stream>>>(
            A, Wt, bias, Cf, Cbf, M, Nn, Kd, relu);
    };
    auto gemmKV = [&](const unsigned short* hE, const float* nA, const float* nB,
                      const float* nC, const unsigned short* Wt, unsigned short* Cbf,
                      int Kd, int dec) {
        gemm_kv<<<dim3(RE / 128, HH / 64), dim3(256), 0, stream>>>(
            hE, nA, nB, nC, Eidx, mask, Wt, Cbf, Kd, dec);
    };

    // ---- features ----
    dist_topk<<<dim3(RV), dim3(128), 0, stream>>>(X, mask, Eidx, Dnb);
    dihedral<<<dim3((BB * 3 * NN + 255) / 256), dim3(256), 0, stream>>>(X, V6);
    edge_feat<<<dim3((RE + 255) / 256), dim3(256), 0, stream>>>(Eidx, Dnb, Efeat);
    gemm(Efeat, Wfe_t, bfe, Ep, nullptr, RE, HH, 32, 0);
    ln128<<<dim3(RE), dim3(32), 0, stream>>>(Ep, nullptr, nullptr, nullptr, Kp); // Eln -> Kp
    gemm(Kp, We_t, be, nullptr, hEbf, RE, HH, HH, 0);                             // h_E (bf16)
    noderaw<<<dim3((RV * HH + 255) / 256), dim3(256), 0, stream>>>(V6, Wfn, bfn, vraw);
    ln128<<<dim3(RV), dim3(32), 0, stream>>>(vraw, nullptr, nullptr, nullptr, attno); // Vln
    gemm(attno, Wv_t, bv, hV, hVbf, RV, HH, HH, 0);                               // h_V
    gather_hs<<<dim3((RV * HH + 255) / 256), dim3(256), 0, stream>>>(S, Ws_, hS);

    // ---- encoder ----
    for (int l = 0; l < 3; ++l) {
        gemm(hVbf, eWQt[l], nullptr, Qb, nullptr, RV, HH, HH, 0);
        gemmKV(hEbf, hV, nullptr, nullptr, eWKt[l], Kp, 2 * HH, 0);
        gemmKV(hEbf, hV, nullptr, nullptr, eWVt[l], Vp, 2 * HH, 0);
        attn<<<dim3(RV), dim3(128), 0, stream>>>(Qb, Kp, Vp, Eidx, mask, attno, 1);
        gemm(attno, eWOt[l], nullptr, dhb, nullptr, RV, HH, HH, 0);
        ln128<<<dim3(RV), dim3(32), 0, stream>>>(hV, dhb, nullptr, hV, hVbf);
        gemm(hVbf, eW1t[l], eb1 + (size_t)l * 4 * HH, nullptr, ffn, RV, 4 * HH, HH, 1);
        gemm(ffn, eW2t[l], eb2 + (size_t)l * HH, dhb, nullptr, RV, HH, 4 * HH, 0);
        ln128<<<dim3(RV), dim3(32), 0, stream>>>(hV, dhb, mask, hV, hVbf);
    }
    hipMemcpyAsync(hVenc, hV, (size_t)RV * HH * 4, hipMemcpyDeviceToDevice, stream);
    scale_hE<<<dim3((RE * HH + 255) / 256), dim3(256), 0, stream>>>(hEbf, mask, hEmask);

    // ---- decoder ----
    for (int l = 0; l < 3; ++l) {
        gemm(hVbf, dWQt[l], nullptr, Qb, nullptr, RV, HH, HH, 0);
        gemmKV(hEmask, hS, hV, hVenc, dWKt[l], Kp, 3 * HH, 1);
        gemmKV(hEmask, hS, hV, hVenc, dWVt[l], Vp, 3 * HH, 1);
        attn<<<dim3(RV), dim3(128), 0, stream>>>(Qb, Kp, Vp, Eidx, mask, attno, 0);
        gemm(attno, dWOt[l], nullptr, dhb, nullptr, RV, HH, HH, 0);
        ln128<<<dim3(RV), dim3(32), 0, stream>>>(hV, dhb, nullptr, hV, hVbf);
        gemm(hVbf, dW1t[l], db1 + (size_t)l * 4 * HH, nullptr, ffn, RV, 4 * HH, HH, 1);
        gemm(ffn, dW2t[l], db2 + (size_t)l * HH, dhb, nullptr, RV, HH, 4 * HH, 0);
        ln128<<<dim3(RV), dim3(32), 0, stream>>>(hV, dhb, mask, hV, hVbf);
    }

    out_head<<<dim3(RV), dim3(32), 0, stream>>>(hV, Wout, bout, (float*)d_out);
}
